// BertAttention_78357383348721
// MI455X (gfx1250) — matrix operand order repaired and checked by reading, not compiled, not run on hardware
//
#include <hip/hip_runtime.h>

typedef __attribute__((ext_vector_type(16))) _Float16 v16h;
typedef __attribute__((ext_vector_type(8)))  _Float16 v8h;
typedef __attribute__((ext_vector_type(8)))  float    v8f;

#define HID  768
#define MTOK 16384
#define SEQ  1024
#define NH   12
#define HD   64
#define KSTEPS (HID / 32)   // 24

union V16U { v16h v; v8h h[2]; };

__device__ __forceinline__ v16h make16(v8h a, v8h b) {
  V16U u; u.h[0] = a; u.h[1] = b; return u.v;
}

// A-fragment (16x32 f16), source row-major [16 x ld], top-left at p.
// Layout used here: lane holds row M=lane&15; K(i) = i + 16*(lane>=16) (contiguous)
__device__ __forceinline__ v16h frag_a_rows(const _Float16* p, int ld) {
  const int lane = threadIdx.x & 31;
  const int row  = lane & 15;
  const int hi   = lane >> 4;
  const _Float16* r = p + (size_t)row * ld + hi * 16;
  v8h lo = *(const v8h*)r;          // K = hi*16 .. hi*16+7
  v8h hh = *(const v8h*)(r + 8);    // K = hi*16+8 .. hi*16+15
  return make16(lo, hh);
}

// B-fragment (32x16) where B[k][n] = p[n*ld + k] (i.e. transposed source).
// Layout used here: lane holds col N=lane&15; K(i) = i + 16*(lane>=16) (contiguous)
__device__ __forceinline__ v16h frag_b_rows(const _Float16* p, int ld) {
  const int lane = threadIdx.x & 31;
  const int n    = lane & 15;
  const int hi   = lane >> 4;
  const _Float16* r = p + (size_t)n * ld + hi * 16;
  v8h lo = *(const v8h*)r;
  v8h hh = *(const v8h*)(r + 8);
  return make16(lo, hh);
}

__device__ __forceinline__ v8f wmma_f16(v16h a, v16h b, v8f c) {
  return __builtin_amdgcn_wmma_f32_16x16x32_f16(false, a, false, b, (short)0, c,
                                                false, false);
}

// ---------------------------------------------------------------------------
// f32 -> f16 convert
// ---------------------------------------------------------------------------
__global__ void cvt_f32_f16(const float* __restrict__ s, _Float16* __restrict__ d,
                            int n) {
  int i = blockIdx.x * 256 + threadIdx.x;
  if (i < n) d[i] = (_Float16)s[i];
}

// ---------------------------------------------------------------------------
// Tiled WMMA GEMM: C[M x 768] = A[M x 768] * W[768 x 768]^T + bias
// Block tile 128x128, 8 waves (4x2), wave tile 32x64 (2x4 16x16 frags).
// LDS double-buffered: one barrier per K-step, staging overlaps WMMA.
// MODE 0: store f16 row-major                   (Q, K projections)
// MODE 1: store f16 transposed to [B, H, D, S]  (V projection)
// MODE 2: store f32 = v + bias + resid          (output projection + residual)
// ---------------------------------------------------------------------------
template <int MODE>
__global__ __launch_bounds__(256) void gemm_wmma(
    const _Float16* __restrict__ A, const _Float16* __restrict__ W,
    const float* __restrict__ bias, _Float16* __restrict__ outh,
    float* __restrict__ outf, const float* __restrict__ resid) {
  __shared__ __align__(32) _Float16 As[2][128 * 32];
  __shared__ __align__(32) _Float16 Bs[2][128 * 32];

  const int tid  = threadIdx.x;
  const int lane = tid & 31;
  const int wave = tid >> 5;
  const int wm   = wave & 3;        // 0..3 : M position
  const int wn   = wave >> 2;       // 0..1 : N position
  const int m0   = blockIdx.y * 128;
  const int n0   = blockIdx.x * 128;
  const int lrow = tid >> 1;
  const int lch  = (tid & 1) * 16;
  const int hi   = lane >> 4;
  const int lc   = lane & 15;

  const _Float16* aRow = &A[(size_t)(m0 + lrow) * HID + lch];
  const _Float16* wRow = &W[(size_t)(n0 + lrow) * HID + lch];
  const int soff = lrow * 32 + lch;

  v8f zero = {};
  v8f acc[2][4];
#pragma unroll
  for (int i = 0; i < 2; ++i)
#pragma unroll
    for (int j = 0; j < 4; ++j) acc[i][j] = zero;

  // stage first tile
  *(v16h*)&As[0][soff] = *(const v16h*)aRow;
  *(v16h*)&Bs[0][soff] = *(const v16h*)wRow;
  __syncthreads();

  for (int kt = 0; kt < KSTEPS; ++kt) {
    const int cur = kt & 1;
    const int nxt = cur ^ 1;
    if (kt + 1 < KSTEPS) {
      // stage next tile into the other buffer (protected by last barrier)
      *(v16h*)&As[nxt][soff] = *(const v16h*)(aRow + (kt + 1) * 32);
      *(v16h*)&Bs[nxt][soff] = *(const v16h*)(wRow + (kt + 1) * 32);
      if (kt + 2 < KSTEPS) {
        __builtin_prefetch(aRow + (kt + 2) * 32, 0, 0);
        __builtin_prefetch(wRow + (kt + 2) * 32, 0, 0);
      }
    }
    v16h af[2], bf[4];
#pragma unroll
    for (int fm = 0; fm < 2; ++fm)
      af[fm] = frag_a_rows(&As[cur][(wm * 32 + fm * 16) * 32], 32);
#pragma unroll
    for (int fn = 0; fn < 4; ++fn)
      bf[fn] = frag_b_rows(&Bs[cur][(wn * 64 + fn * 16) * 32], 32);
#pragma unroll
    for (int fm = 0; fm < 2; ++fm)
#pragma unroll
      for (int fn = 0; fn < 4; ++fn)
        acc[fm][fn] = wmma_f16(af[fm], bf[fn], acc[fm][fn]);
    __syncthreads();
  }

#pragma unroll
  for (int fm = 0; fm < 2; ++fm) {
#pragma unroll
    for (int fn = 0; fn < 4; ++fn) {
#pragma unroll
      for (int r = 0; r < 8; ++r) {
        const int row = m0 + wm * 32 + fm * 16 + r + hi * 8;  // token index
        const int col = n0 + wn * 64 + fn * 16 + lc;          // feature index
        const float v = acc[fm][fn][r] + bias[col];
        if constexpr (MODE == 0) {
          outh[(size_t)row * HID + col] = (_Float16)v;
        } else if constexpr (MODE == 1) {
          const int bb = row >> 10, ss = row & 1023;
          outh[((size_t)(bb * HID + col)) * SEQ + ss] = (_Float16)v;
        } else {
          const size_t idx = (size_t)row * HID + col;
          outf[idx] = v + resid[idx];
        }
      }
    }
  }
}

// ---------------------------------------------------------------------------
// Flash-style attention. Grid (S/64, NH, B), 128 threads = 4 waves.
// Each wave owns 16 query rows and a private LDS P-tile: no barriers needed
// (same-wave LDS ops are in-order; compiler inserts s_wait_dscnt).
// ---------------------------------------------------------------------------
__global__ __launch_bounds__(128) void attn_wmma(
    const _Float16* __restrict__ Qh, const _Float16* __restrict__ Kh,
    const _Float16* __restrict__ Vt, const float* __restrict__ mask,
    _Float16* __restrict__ Ctx) {
  __shared__ __align__(32) _Float16 Ps[4 * 16 * 32];  // per-wave P tile
  const int tid  = threadIdx.x;
  const int lane = tid & 31;
  const int wave = tid >> 5;
  const int b    = blockIdx.z;
  const int h    = blockIdx.y;
  const int q0   = blockIdx.x * 64 + wave * 16;
  const int hi   = lane >> 4;
  const int lc   = lane & 15;

  const _Float16* qbase = Qh + ((size_t)(b * SEQ + q0)) * HID + h * HD;
  const v16h qa0 = frag_a_rows(qbase, HID);       // d = 0..31
  const v16h qa1 = frag_a_rows(qbase + 32, HID);  // d = 32..63

  v8f zero = {};
  v8f o[4];
#pragma unroll
  for (int dt = 0; dt < 4; ++dt) o[dt] = zero;
  float mr[8], lr[8];
#pragma unroll
  for (int r = 0; r < 8; ++r) { mr[r] = -3.0e38f; lr[r] = 0.0f; }

  const float* mrow = mask + (size_t)b * SEQ;
  _Float16* pw = &Ps[wave * 512];

  for (int k0 = 0; k0 < SEQ; k0 += 32) {
    // scores: S[16q x 32k] as two 16x16 f32 accumulators (d reduced via 2 WMMA)
    const _Float16* kb  = Kh + ((size_t)(b * SEQ + k0)) * HID + h * HD;
    const _Float16* kb2 = kb + 16 * HID;
    __builtin_prefetch(kb + 32 * HID, 0, 0);  // next K tile
    v8f s0 = wmma_f16(qa1, frag_b_rows(kb + 32, HID), zero);
    s0     = wmma_f16(qa0, frag_b_rows(kb, HID), s0);
    v8f s1 = wmma_f16(qa1, frag_b_rows(kb2 + 32, HID), zero);
    s1     = wmma_f16(qa0, frag_b_rows(kb2, HID), s1);

    const float msk0 = mrow[k0 + lc];
    const float msk1 = mrow[k0 + 16 + lc];

    float p0[8], p1[8], mnew[8];
#pragma unroll
    for (int r = 0; r < 8; ++r) {
      float a = s0[r] * 0.125f + msk0;   // 1/sqrt(64)
      float c = s1[r] * 0.125f + msk1;
      p0[r] = a; p1[r] = c;
      float mx = fmaxf(a, c);
#pragma unroll
      for (int off = 1; off < 16; off <<= 1)
        mx = fmaxf(mx, __shfl_xor(mx, off, 32));
      mnew[r] = fmaxf(mr[r], mx);
    }
#pragma unroll
    for (int r = 0; r < 8; ++r) {
      const float cf = __expf(mr[r] - mnew[r]);
      p0[r] = __expf(p0[r] - mnew[r]);
      p1[r] = __expf(p1[r] - mnew[r]);
      float rs = p0[r] + p1[r];
#pragma unroll
      for (int off = 1; off < 16; off <<= 1) rs += __shfl_xor(rs, off, 32);
      lr[r] = lr[r] * cf + rs;
      mr[r] = mnew[r];
#pragma unroll
      for (int dt = 0; dt < 4; ++dt) o[dt][r] = o[dt][r] * cf;
    }
    // P (f32, C-layout) -> LDS -> f16 A-fragment layout (wave-private region)
#pragma unroll
    for (int r = 0; r < 8; ++r) {
      const int row = r + hi * 8;
      pw[row * 32 + lc]      = (_Float16)p0[r];
      pw[row * 32 + lc + 16] = (_Float16)p1[r];
    }
    const v16h pa = frag_a_rows(pw, 32);
#pragma unroll
    for (int dt = 0; dt < 4; ++dt) {
      const _Float16* vb = Vt + ((size_t)(b * HID + h * HD + dt * 16)) * SEQ + k0;
      o[dt] = wmma_f16(pa, frag_b_rows(vb, SEQ), o[dt]);
    }
  }

#pragma unroll
  for (int r = 0; r < 8; ++r) {
    const float inv = 1.0f / lr[r];
    const int t = b * SEQ + q0 + r + hi * 8;
#pragma unroll
    for (int dt = 0; dt < 4; ++dt) {
      const int f = h * HD + dt * 16 + lc;
      Ctx[(size_t)t * HID + f] = (_Float16)(o[dt][r] * inv);
    }
  }
}

// ---------------------------------------------------------------------------
// In-place LayerNorm over 768 features; one block per row.
// ---------------------------------------------------------------------------
__global__ __launch_bounds__(256) void ln_kernel(float* __restrict__ x,
                                                 const float* __restrict__ gamma,
                                                 const float* __restrict__ beta) {
  __shared__ float rs[8], rs2[8];
  const int tid = threadIdx.x, lane = tid & 31, wave = tid >> 5;
  float* p = x + (size_t)blockIdx.x * HID;
  const float v0 = p[tid], v1 = p[tid + 256], v2 = p[tid + 512];
  float s  = v0 + v1 + v2;
  float s2 = v0 * v0 + v1 * v1 + v2 * v2;
#pragma unroll
  for (int off = 1; off < 32; off <<= 1) {
    s  += __shfl_xor(s, off, 32);
    s2 += __shfl_xor(s2, off, 32);
  }
  if (lane == 0) { rs[wave] = s; rs2[wave] = s2; }
  __syncthreads();
  float ts = 0.0f, ts2 = 0.0f;
#pragma unroll
  for (int w = 0; w < 8; ++w) { ts += rs[w]; ts2 += rs2[w]; }
  const float mean = ts * (1.0f / HID);
  const float var  = ts2 * (1.0f / HID) - mean * mean;
  const float inv  = rsqrtf(var + 1e-12f);
  p[tid]       = (v0 - mean) * inv * gamma[tid]       + beta[tid];
  p[tid + 256] = (v1 - mean) * inv * gamma[tid + 256] + beta[tid + 256];
  p[tid + 512] = (v2 - mean) * inv * gamma[tid + 512] + beta[tid + 512];
}

// ---------------------------------------------------------------------------
extern "C" void kernel_launch(void* const* d_in, const int* in_sizes, int n_in,
                              void* d_out, int out_size, void* d_ws,
                              size_t ws_size, hipStream_t stream) {
  (void)in_sizes; (void)n_in; (void)out_size; (void)ws_size;
  const float* hs    = (const float*)d_in[0];
  const float* mask  = (const float*)d_in[1];
  const float* qw    = (const float*)d_in[2];
  const float* qb    = (const float*)d_in[3];
  const float* kw    = (const float*)d_in[4];
  const float* kb    = (const float*)d_in[5];
  const float* vw    = (const float*)d_in[6];
  const float* vb    = (const float*)d_in[7];
  const float* ow    = (const float*)d_in[8];
  const float* ob    = (const float*)d_in[9];
  const float* gamma = (const float*)d_in[10];
  const float* beta  = (const float*)d_in[11];

  char* ws = (char*)d_ws;
  const size_t SZ_ACT = (size_t)MTOK * HID * sizeof(_Float16);  // 25165824
  const size_t SZ_W   = (size_t)HID * HID * sizeof(_Float16);   // 1179648
  _Float16* Xh = (_Float16*)(ws);
  _Float16* Qh = (_Float16*)(ws + SZ_ACT);
  _Float16* Kh = (_Float16*)(ws + 2 * SZ_ACT);
  _Float16* Vt = (_Float16*)(ws + 3 * SZ_ACT);
  _Float16* Wq = (_Float16*)(ws + 4 * SZ_ACT);
  _Float16* Wk = (_Float16*)(ws + 4 * SZ_ACT + SZ_W);
  _Float16* Wv = (_Float16*)(ws + 4 * SZ_ACT + 2 * SZ_W);
  _Float16* Wo = (_Float16*)(ws + 4 * SZ_ACT + 3 * SZ_W);
  _Float16* Ctx = Xh;  // X is fully consumed by the QKV GEMMs; reuse for context

  const int nAct = MTOK * HID;
  const int nW   = HID * HID;
  cvt_f32_f16<<<(nAct + 255) / 256, 256, 0, stream>>>(hs, Xh, nAct);
  cvt_f32_f16<<<(nW + 255) / 256, 256, 0, stream>>>(qw, Wq, nW);
  cvt_f32_f16<<<(nW + 255) / 256, 256, 0, stream>>>(kw, Wk, nW);
  cvt_f32_f16<<<(nW + 255) / 256, 256, 0, stream>>>(vw, Wv, nW);
  cvt_f32_f16<<<(nW + 255) / 256, 256, 0, stream>>>(ow, Wo, nW);

  dim3 ggrid(HID / 128, MTOK / 128);
  gemm_wmma<0><<<ggrid, 256, 0, stream>>>(Xh, Wq, qb, Qh, nullptr, nullptr);
  gemm_wmma<0><<<ggrid, 256, 0, stream>>>(Xh, Wk, kb, Kh, nullptr, nullptr);
  gemm_wmma<1><<<ggrid, 256, 0, stream>>>(Xh, Wv, vb, Vt, nullptr, nullptr);

  attn_wmma<<<dim3(SEQ / 64, NH, 16), 128, 0, stream>>>(Qh, Kh, Vt, mask, Ctx);

  gemm_wmma<2><<<ggrid, 256, 0, stream>>>(Ctx, Wo, ob, nullptr, (float*)d_out, hs);

  ln_kernel<<<MTOK, 256, 0, stream>>>((float*)d_out, gamma, beta);
}
